// PathwayPredictor_63797444214829
// MI455X (gfx1250) — matrix operand order repaired and checked by reading, not compiled, not run on hardware
//
#include <hip/hip_runtime.h>

// ---------------------------------------------------------------------------
// MPNN pathway predictor for MI455X (gfx1250, wave32, WMMA + async-to-LDS).
//
// Roofline: 19.7 GFLOP vs 134 MB streamed adjacency -> ~6us floor at
// 23.3 TB/s. bf16 WMMA (16x16x32, f32 accum) matches compute to that floor.
// Prologue kernels pre-convert weights + embed table to bf16 LDS images in
// d_ws (L2-resident), so per-block staging is pure GLOBAL_LOAD_ASYNC_TO_LDS
// B128 copies (ASYNCcnt) instead of per-element convert loops.
// One workgroup (4 wave32) per molecule; each wave owns a 16-atom M-tile.
// ---------------------------------------------------------------------------

typedef __bf16 bf16x16 __attribute__((ext_vector_type(16)));
typedef __bf16 bf16x2  __attribute__((ext_vector_type(2)));
typedef float  f32x8   __attribute__((ext_vector_type(8)));
typedef float  f32x4   __attribute__((ext_vector_type(4)));

union BfVec { bf16x16 v; unsigned int u[8]; };

#define XS 104   // X   row stride (bf16), rows = atoms(64), cols = feat (96 used)
#define WS 104   // Wt  row stride, rows = out-feat n(80), cols = in-feat k (96 used)
#define HS 72    // Ht  row stride, rows = feat n(80),   cols = atom m (64 used)
#define AS 72    // Adj row stride, rows = atom m(64),   cols = atom k (64 used)

// d_ws byte layout
#define WS_WT1   0         // bf16 [80][104] image of W1^T (16640 B)
#define WS_WT2   16640     // bf16 [80][104] image of W2^T
#define WS_BIAS1 33280     // f32 [96]
#define WS_BIAS2 33664     // f32 [96]
#define WS_EMB   34048     // bf16 [10000][96], rows 192 B (1.92 MB)

__device__ __forceinline__ unsigned short bfbits(float f) {
  union { __bf16 h; unsigned short s; } t; t.h = (__bf16)f; return t.s;
}
__device__ __forceinline__ unsigned packbf2(float a, float b) {
  union { bf16x2 v; unsigned u; } t;
  t.v.x = (__bf16)a; t.v.y = (__bf16)b; return t.u;
}
// CDNA5 async copy: 16B global -> LDS, tracked by ASYNCcnt
__device__ __forceinline__ void async_b128(unsigned lds, const void* g) {
  asm volatile("global_load_async_to_lds_b128 %0, %1, off"
               :: "v"(lds), "v"((unsigned long long)g) : "memory");
}
__device__ __forceinline__ void wait_async0() {
  asm volatile("s_wait_asynccnt 0" ::: "memory");
}

// ======================= prologue: weight images ===========================
__global__ __launch_bounds__(256) void
prep_weights(const float* __restrict__ w1, const float* __restrict__ b1,
             const float* __restrict__ w2, const float* __restrict__ b2,
             char* __restrict__ wsb) {
  int j = blockIdx.x * 256 + threadIdx.x;
  unsigned short* wt1 = (unsigned short*)(wsb + WS_WT1);
  unsigned short* wt2 = (unsigned short*)(wsb + WS_WT2);
  if (j < 80 * WS) {
    int n = j / WS, k = j - n * WS;
    bool valid = (n < 70) & (k < 70);
    wt1[j] = bfbits(valid ? w1[k * 70 + n] : 0.f);
    wt2[j] = bfbits(valid ? w2[k * 70 + n] : 0.f);
  }
  if (blockIdx.x == 0 && threadIdx.x < 96) {
    int t = threadIdx.x;
    ((float*)(wsb + WS_BIAS1))[t] = (t < 70) ? b1[t] : 0.f;
    ((float*)(wsb + WS_BIAS2))[t] = (t < 70) ? b2[t] : 0.f;
  }
}

// ======================= prologue: embed table bf16 ========================
__global__ __launch_bounds__(256) void
prep_embed(const float* __restrict__ emb, char* __restrict__ wsb) {
  int i = blockIdx.x * 256 + threadIdx.x;
  if (i < 10000 * 96) {
    int row = i / 96, d = i - row * 96;
    ((unsigned short*)(wsb + WS_EMB))[i] = bfbits(d < 70 ? emb[row * 70 + d] : 0.f);
  }
}

// ============================ main kernel ==================================
__global__ __launch_bounds__(128) void
pathway_mpnn_kernel(const int*   __restrict__ atoms,   // [B,64]
                    const float* __restrict__ adj,     // [B,64,64]
                    const float* __restrict__ sel,     // [B,7]
                    const float* __restrict__ wp,      // [77,11]
                    const float* __restrict__ bp,      // [11]
                    const char*  __restrict__ wsb,     // prepped images
                    float*       __restrict__ out)     // [B,11]
{
  const int b    = blockIdx.x;
  const int tid  = threadIdx.x;
  const int wave = tid >> 5;
  const int lane = tid & 31;
  const int half = lane >> 4;
  const int nl   = lane & 15;

  __shared__ __align__(16) unsigned short Xs  [64 * XS];
  __shared__ __align__(16) unsigned short Wts [80 * WS];
  __shared__ __align__(16) unsigned short Hs  [80 * HS];
  __shared__ __align__(16) unsigned short Adjs[64 * AS];
  __shared__ float biasS[96];
  __shared__ float ysum [96];

  const float* bias1 = (const float*)(wsb + WS_BIAS1);
  const float* bias2 = (const float*)(wsb + WS_BIAS2);
  const char*  embB  = wsb + WS_EMB;

  if (tid < 96) { ysum[tid] = 0.f; biasS[tid] = bias1[tid]; }

  const unsigned XsB = (unsigned)(unsigned long long)&Xs[0];
  const unsigned WtB = (unsigned)(unsigned long long)&Wts[0];

  // ---- embeddings: async-copy 64 pre-converted bf16 rows (12 x 16B each)
  {
    const int* arow = atoms + b * 64;
#pragma unroll
    for (int it = 0; it < 6; ++it) {
      int c = tid + 128 * it;          // 0..767 row-chunks
      int row = c / 12, cc = c - row * 12;
      async_b128(XsB + row * (XS * 2) + cc * 16,
                 embB + arow[row] * 192 + cc * 16);
    }
  }
  // ---- W1 image: linear async copy of [80][104] bf16 (1040 x 16B)
#pragma unroll
  for (int it = 0; it < 9; ++it) {
    int c = tid + 128 * it;
    if (c < 1040) async_b128(WtB + c * 16, wsb + WS_WT1 + c * 16);
  }
  // ---- adjacency: stream f32 (non-temporal), convert to bf16 in LDS
  {
    const f32x4* a4 = (const f32x4*)(adj + (size_t)b * 4096);
    for (int i = tid; i < 1024; i += 128) {
      f32x4 v = __builtin_nontemporal_load(&a4[i]);
      int row = i >> 4, col = (i & 15) * 4;
      *(unsigned int*)&Adjs[row * AS + col]     = packbf2(v.x, v.y);
      *(unsigned int*)&Adjs[row * AS + col + 2] = packbf2(v.z, v.w);
    }
  }
  wait_async0();
  __syncthreads();

  const int m  = wave * 16 + nl;        // A-operand row (atom) for this lane
  const int m0 = wave * 16 + half * 8;  // C-tile row base for this lane

  for (int l = 0; l < 2; ++l) {
    // ============ matmul1: Ht = relu(X @ W + b), K = 96 (3 steps) ==========
    for (int nt = 0; nt < 5; ++nt) {
      const int n = nt * 16 + nl;
      float bv = biasS[n];
      f32x8 acc = {bv, bv, bv, bv, bv, bv, bv, bv};
#pragma unroll
      for (int ks = 0; ks < 3; ++ks) {
        BfVec Aop, Bop;
        const unsigned short* arow = &Xs[m * XS + ks * 32 + 8 * half];
#pragma unroll
        for (int p = 0; p < 4; ++p) {       // A 16x32 layout (ISA 7.12.2)
          Aop.u[p]     = *(const unsigned int*)(arow + 2 * p);
          Aop.u[p + 4] = *(const unsigned int*)(arow + 16 + 2 * p);
        }
        const unsigned short* brow = &Wts[n * WS + ks * 32 + 8 * half];
#pragma unroll
        for (int p = 0; p < 4; ++p) {       // B 32x16 layout
          Bop.u[p]     = *(const unsigned int*)(brow + 2 * p);
          Bop.u[p + 4] = *(const unsigned int*)(brow + 16 + 2 * p);
        }
        acc = __builtin_amdgcn_wmma_f32_16x16x32_bf16(
            false, Aop.v, false, Bop.v, (short)0, acc, false, false);
      }
#pragma unroll
      for (int r = 0; r < 8; r += 2)        // relu + pack, store Ht[n][m]
        *(unsigned int*)&Hs[n * HS + m0 + r] =
            packbf2(fmaxf(acc[r], 0.f), fmaxf(acc[r + 1], 0.f));
    }
    __syncthreads();

    if (l == 0) {      // prefetch W2 image + bias2, overlapped with matmul2
#pragma unroll
      for (int it = 0; it < 9; ++it) {
        int c = tid + 128 * it;
        if (c < 1040) async_b128(WtB + c * 16, wsb + WS_WT2 + c * 16);
      }
      if (tid < 96) biasS[tid] = bias2[tid];
    }

    // ============ matmul2: X' = Adj @ Ht, K = 64 (2 steps) =================
    for (int nt = 0; nt < 5; ++nt) {
      const int n = nt * 16 + nl;
      f32x8 acc = {0.f, 0.f, 0.f, 0.f, 0.f, 0.f, 0.f, 0.f};
#pragma unroll
      for (int ks = 0; ks < 2; ++ks) {
        BfVec Aop, Bop;
        const unsigned short* arow = &Adjs[m * AS + ks * 32 + 8 * half];
#pragma unroll
        for (int p = 0; p < 4; ++p) {
          Aop.u[p]     = *(const unsigned int*)(arow + 2 * p);
          Aop.u[p + 4] = *(const unsigned int*)(arow + 16 + 2 * p);
        }
        const unsigned short* brow = &Hs[n * HS + ks * 32 + 8 * half];
#pragma unroll
        for (int p = 0; p < 4; ++p) {
          Bop.u[p]     = *(const unsigned int*)(brow + 2 * p);
          Bop.u[p + 4] = *(const unsigned int*)(brow + 16 + 2 * p);
        }
        acc = __builtin_amdgcn_wmma_f32_16x16x32_bf16(
            false, Aop.v, false, Bop.v, (short)0, acc, false, false);
      }
      if (l == 0) {
#pragma unroll
        for (int r = 0; r < 8; ++r)         // becomes next layer's X
          Xs[(m0 + r) * XS + n] = bfbits(acc[r]);
      } else {
        float s = acc[0] + acc[1] + acc[2] + acc[3] +
                  acc[4] + acc[5] + acc[6] + acc[7];
        atomicAdd(&ysum[n], s);             // ds_add_f32 pool over atoms
      }
    }
    if (l == 0) wait_async0();              // W2 image landed before reuse
    __syncthreads();
  }

  // ---- final projection: out[c] = [ysum, sel] @ wp + bp
  if (tid < 11) {
    float acc = bp[tid];
    for (int d = 0; d < 70; ++d) acc += ysum[d] * wp[d * 11 + tid];
    for (int e = 0; e < 7; ++e)  acc += sel[b * 7 + e] * wp[(70 + e) * 11 + tid];
    out[b * 11 + tid] = acc;
  }
}

extern "C" void kernel_launch(void* const* d_in, const int* in_sizes, int n_in,
                              void* d_out, int out_size, void* d_ws, size_t ws_size,
                              hipStream_t stream) {
  (void)in_sizes; (void)n_in; (void)ws_size; (void)out_size;
  const int*   atoms = (const int*)  d_in[0];
  const float* adjm  = (const float*)d_in[1];
  const float* sel   = (const float*)d_in[2];
  const float* embed = (const float*)d_in[3];
  const float* w1    = (const float*)d_in[4];
  const float* b1    = (const float*)d_in[5];
  const float* w2    = (const float*)d_in[6];
  const float* b2    = (const float*)d_in[7];
  const float* wp    = (const float*)d_in[8];
  const float* bp    = (const float*)d_in[9];
  char*  wsb = (char*)d_ws;
  float* out = (float*)d_out;

  prep_weights<<<(80 * WS + 255) / 256, 256, 0, stream>>>(w1, b1, w2, b2, wsb);
  prep_embed<<<(10000 * 96 + 255) / 256, 256, 0, stream>>>(embed, wsb);
  pathway_mpnn_kernel<<<8192, 128, 0, stream>>>(atoms, adjm, sel, wp, bp, wsb, out);
}
